// Quantizer_71193377899422
// MI455X (gfx1250) — matrix operand order repaired and checked by reading, not compiled, not run on hardware
//
#include <hip/hip_runtime.h>
#include <hip/hip_bf16.h>

typedef __attribute__((ext_vector_type(16))) _Float16 v16h;
typedef __attribute__((ext_vector_type(4)))  _Float16 v4h;
typedef __attribute__((ext_vector_type(8)))  float    v8f;

#define K_CAT 1024
#define DIM   64
#define HW    4096              // 64*64
#define NB    16
#define NTOT  (NB * HW)         // 65536 vectors
#define NOUT  (NTOT * DIM)      // 4194304 output elements
#define WPB   4                 // waves per block in argmin kernel
#define TPW   4                 // 16-row M-tiles per wave (B-operand reuse)

// ws layout (bytes):
//   [0,       131072)  embT  : _Float16 [DIM][K_CAT]   (transposed fp16 codebook)
//   [131072,  135168)  enorm : float    [K_CAT]        (||e_k||^2)
//   [135168,  397312)  ids   : int      [NTOT]
//   [397312,  397316)  acc   : float                   (loss accumulator)
// NOTE: the argmin kernel's software prefetch intentionally over-reads up to
// 64B past embT/enorm; those reads land inside later ws regions (in-bounds).
#define WS_ENORM_OFF 131072
#define WS_IDS_OFF   135168
#define WS_ACC_OFF   397312

// --- Kernel 1: codebook transpose to fp16 + squared norms (+ zero acc) -----
__global__ __launch_bounds__(64) void vq_prep(const float* __restrict__ emb,
                                              _Float16* __restrict__ embT,
                                              float* __restrict__ enorm,
                                              float* __restrict__ acc) {
    int k = blockIdx.x * 64 + threadIdx.x;
    if (k == 0) *acc = 0.0f;
    if (k >= K_CAT) return;
    float s = 0.0f;
    for (int c = 0; c < DIM; ++c) {
        float v = emb[(size_t)k * DIM + c];
        s += v * v;
        embT[(size_t)c * K_CAT + k] = (_Float16)v;
    }
    enorm[k] = s;
}

// --- Kernel 2: WMMA distance GEMM + fused argmin ---------------------------
// Each wave handles TPW 16-row tiles of ze against all 1024 codes, reusing
// each B tile (codebook slice) across TPW accumulator sets. B tiles and the
// ||e||^2 terms are double-buffered so next-iteration loads overlap WMMAs.
__global__ __launch_bounds__(32 * WPB) void vq_argmin(
    const float* __restrict__ x, const _Float16* __restrict__ embT,
    const float* __restrict__ enorm, int* __restrict__ ids) {
    // Transposed ze tiles in LDS: ztT[c][r], row stride padded to 24 halves.
    __shared__ __align__(16) _Float16 ztT[WPB][TPW][DIM][24];

    const int lane  = threadIdx.x & 31;
    const int wv    = threadIdx.x >> 5;
    const int tile0 = (blockIdx.x * WPB + wv) * TPW;

    // --- Stage TPW tiles: float4 loads along contiguous hw, fp16 v4h stores.
    {
        const int c8 = lane >> 2;        // 0..7
        const int r4 = (lane & 3) * 4;   // 0,4,8,12
        #pragma unroll
        for (int t = 0; t < TPW; ++t) {
            const int nbase = (tile0 + t) * 16;
            const int b  = nbase >> 12;
            const int hw = nbase & 4095;
            #pragma unroll
            for (int i = 0; i < 8; ++i) {
                int c = i * 8 + c8;
                float4 v = *(const float4*)&x[((size_t)(b * DIM + c) << 12) + hw + r4];
                v4h h = { (_Float16)v.x, (_Float16)v.y, (_Float16)v.z, (_Float16)v.w };
                *(v4h*)&ztT[wv][t][c][r4] = h;
            }
        }
    }
    __syncthreads();

    // --- Build A operands per 16-bit 16x32 A layout:
    // lane<16: halves j=0..7 -> K=j, j=8..15 -> K=8+j; lane>=16: +8.
    const int row = lane & 15;
    const int hiK = (lane >> 4) << 3;    // 0 or 8
    v16h a0[TPW], a1[TPW];
    #pragma unroll
    for (int t = 0; t < TPW; ++t) {
        #pragma unroll
        for (int j = 0; j < 16; ++j) {
            int K = j + ((j >> 3) << 3) + hiK;
            a0[t][j] = ztT[wv][t][K][row];
            a1[t][j] = ztT[wv][t][32 + K][row];
        }
    }

    float minv[TPW][8];
    int   mini[TPW][8];
    #pragma unroll
    for (int t = 0; t < TPW; ++t)
        #pragma unroll
        for (int r = 0; r < 8; ++r) { minv[t][r] = 3.4e38f; mini[t][r] = 0; }

    // B operand: lane L holds column n=L%16; halves j follow the A K order.
    const _Float16* bp0 = embT + (lane & 15);                   // c = K
    const _Float16* bp1 = embT + (size_t)32 * K_CAT + (lane & 15); // c = 32 + K
    const float*    ep  = enorm + (lane & 15);

    // Prime the double buffers.
    v16h  b0, b1;
    #pragma unroll
    for (int j = 0; j < 16; ++j) {
        int K = j + ((j >> 3) << 3) + hiK;
        b0[j] = bp0[(size_t)K * K_CAT];
        b1[j] = bp1[(size_t)K * K_CAT];
    }
    float e2 = *ep;

    #pragma unroll 2
    for (int kt = 0; kt < K_CAT / 16; ++kt) {
        // Prefetch next iteration's operands (kt=63 over-reads stay in ws).
        v16h  nb0, nb1;
        #pragma unroll
        for (int j = 0; j < 16; ++j) {
            int K = j + ((j >> 3) << 3) + hiK;
            nb0[j] = bp0[(size_t)K * K_CAT + (kt + 1) * 16];
            nb1[j] = bp1[(size_t)K * K_CAT + (kt + 1) * 16];
        }
        float ne2 = ep[(kt + 1) * 16];

        const int k = kt * 16 + (lane & 15);
        #pragma unroll
        for (int t = 0; t < TPW; ++t) {
            v8f acc = {};
            acc = __builtin_amdgcn_wmma_f32_16x16x32_f16(
                false, a0[t], false, b0, (short)0, acc, false, false);
            acc = __builtin_amdgcn_wmma_f32_16x16x32_f16(
                false, a1[t], false, b1, (short)0, acc, false, false);
            // C/D layout: VGPR r of lane L holds (M = r + 8*(L>=16), N = L%16).
            #pragma unroll
            for (int r = 0; r < 8; ++r) {
                float d = e2 - 2.0f * acc[r];   // ||e||^2 - 2 z.e (||z||^2 const)
                if (d < minv[t][r]) { minv[t][r] = d; mini[t][r] = k; }
            }
        }
        b0 = nb0; b1 = nb1; e2 = ne2;
    }

    // --- Branchless cross-lane argmin over the 16 lanes sharing each M.
    // Pack (dist, idx) into a sortable u64 key; min(key) == (min dist,
    // tie -> smaller idx), matching jnp.argmin first-occurrence semantics.
    #pragma unroll
    for (int t = 0; t < TPW; ++t) {
        const int nbase = (tile0 + t) * 16;
        #pragma unroll
        for (int r = 0; r < 8; ++r) {
            unsigned int fb = __float_as_uint(minv[t][r]);
            fb ^= (unsigned int)(((int)fb) >> 31) | 0x80000000u;  // monotonic map
            unsigned long long key =
                ((unsigned long long)fb << 32) | (unsigned int)mini[t][r];
            #pragma unroll
            for (int m = 1; m <= 8; m <<= 1) {
                unsigned long long o = __shfl_xor(key, m, 32);
                key = (o < key) ? o : key;      // branchless: cmp_lt_u64 + cndmask
            }
            if ((lane & 15) == r)
                ids[nbase + ((lane >> 4) << 3) + r] = (int)(key & 0xffffffffu);
        }
    }
}

// --- Kernel 3: gather zq, transposed store, loss partial sums --------------
__global__ __launch_bounds__(256) void vq_gather(
    const float* __restrict__ x, const float* __restrict__ emb,
    const int* __restrict__ ids, float* __restrict__ out,
    float* __restrict__ acc) {
    __shared__ float red[256];
    int n  = blockIdx.x * 256 + threadIdx.x;
    int b  = n >> 12;
    int hw = n & 4095;
    int id = ids[n];
    const float* e = emb + (size_t)id * DIM;
    float s = 0.0f;
    #pragma unroll
    for (int c4 = 0; c4 < DIM / 4; ++c4) {
        float4 q = *(const float4*)(e + 4 * c4);          // b128 codebook gather
        #pragma unroll
        for (int j = 0; j < 4; ++j) {
            int c = 4 * c4 + j;
            size_t off = ((size_t)(b * DIM + c) << 12) + hw;   // (b,c,h,w) layout
            float qv = (j == 0) ? q.x : (j == 1) ? q.y : (j == 2) ? q.z : q.w;
            float d = x[off] - qv;
            out[off] = qv;                                 // zq_out == zq forward
            s += d * d;
        }
    }
    red[threadIdx.x] = s;
    __syncthreads();
    for (int st = 128; st > 0; st >>= 1) {
        if (threadIdx.x < st) red[threadIdx.x] += red[threadIdx.x + st];
        __syncthreads();
    }
    if (threadIdx.x == 0) atomicAdd(acc, red[0]);
}

// --- Kernel 4: finalize loss = (1 + BETA) * mean ---------------------------
__global__ void vq_finalize(const float* __restrict__ acc,
                            float* __restrict__ out_loss) {
    out_loss[0] = 1.25f * acc[0] / (float)NOUT;
}

extern "C" void kernel_launch(void* const* d_in, const int* in_sizes, int n_in,
                              void* d_out, int out_size, void* d_ws, size_t ws_size,
                              hipStream_t stream) {
    (void)in_sizes; (void)n_in; (void)out_size; (void)ws_size;
    const float* x   = (const float*)d_in[0];
    const float* emb = (const float*)d_in[1];
    float* out = (float*)d_out;

    char*     ws    = (char*)d_ws;
    _Float16* embT  = (_Float16*)ws;
    float*    enorm = (float*)(ws + WS_ENORM_OFF);
    int*      ids   = (int*)(ws + WS_IDS_OFF);
    float*    acc   = (float*)(ws + WS_ACC_OFF);

    vq_prep    <<<dim3(K_CAT / 64),                dim3(64),       0, stream>>>(emb, embT, enorm, acc);
    vq_argmin  <<<dim3((NTOT / 16) / (WPB * TPW)), dim3(32 * WPB), 0, stream>>>(x, embT, enorm, ids);
    vq_gather  <<<dim3(NTOT / 256),                dim3(256),      0, stream>>>(x, emb, ids, out, acc);
    vq_finalize<<<dim3(1),                         dim3(1),        0, stream>>>(acc, out + NOUT);
}
